// GraphTransformerPE_50002009260135
// MI455X (gfx1250) — matrix operand order repaired and checked by reading, not compiled, not run on hardware
//
#include <hip/hip_runtime.h>
#include <hip/hip_bf16.h>
#include <math.h>

typedef __bf16 bf16_t;
typedef __attribute__((ext_vector_type(16))) __bf16 bf16x16;
typedef __attribute__((ext_vector_type(8)))  __bf16 bf16x8;
typedef __attribute__((ext_vector_type(8)))  float  floatx8;

#define N_GRAPHS   16
#define NODES_PER  420
#define NNODES     (N_GRAPHS * NODES_PER)   // 6720
#define DEG        8
#define NEDGES     (NNODES * DEG)           // 53760
#define IN_DIM     2048
#define HD1        2048                     // HEADS*H1
#define D1         1024
#define HEADS1     2
#define H2DIM      64
#define FC1_K      (NODES_PER * H2DIM)      // 26880
#define FC1_N      256
#define FC2_N      128
#define FC3_N      64
#define OUT_DIM    18

// ---------------------------------------------------------------------------
// Positional-embedding add, fp32 -> bf16
// ---------------------------------------------------------------------------
__global__ void k_pe_add(const float* __restrict__ x, const int* __restrict__ conn,
                         const float* __restrict__ node_pe, const float* __restrict__ lobe_pe,
                         const float* __restrict__ lung_pe, bf16_t* __restrict__ hbf) {
    long idx = (long)blockIdx.x * blockDim.x + threadIdx.x;
    if (idx >= (long)NNODES * IN_DIM) return;
    int i = (int)(idx / IN_DIM);
    int j = (int)(idx % IN_DIM);
    int c = conn[i];
    float v = x[idx]
            + node_pe[(long)(i % NODES_PER) * IN_DIM + j]
            + lobe_pe[(long)(c - 1) * IN_DIM + j]
            + lung_pe[(long)(c <= 2 ? 0 : 1) * IN_DIM + j];
    hbf[idx] = (bf16_t)v;
}

// ---------------------------------------------------------------------------
// Weight transpose + convert: W[K][N] f32 -> Wt[N][K] bf16
// ---------------------------------------------------------------------------
__global__ void k_transpose_bf16(const float* __restrict__ W, bf16_t* __restrict__ Wt,
                                 int K, int N) {
    long idx = (long)blockIdx.x * blockDim.x + threadIdx.x;
    if (idx >= (long)K * N) return;
    int k = (int)(idx / N);
    int n = (int)(idx % N);
    Wt[(long)n * K + k] = (bf16_t)W[idx];
}

// fp32 -> bf16 (with relu; idempotent on already-relu'd data)
__global__ void k_relu_bf16(const float* __restrict__ in, bf16_t* __restrict__ out, long n) {
    long i = (long)blockIdx.x * blockDim.x + threadIdx.x;
    if (i >= n) return;
    out[i] = (bf16_t)fmaxf(in[i], 0.0f);
}

// ---------------------------------------------------------------------------
// WMMA fragment helpers (V_WMMA_F32_16X16X32_BF16 operand layouts)
// A: lanes 0-15 rows, K chunks {0..7,16..23}; lanes 16-31 chunks {8..15,24..31}
// B: lane%16 = column, same K chunks as A: lanes<16 {0..7,16..23}, lanes>=16 {8..15,24..31}
// ---------------------------------------------------------------------------
__device__ inline bf16x16 pack16(bf16x8 lo, bf16x8 hi) {
    bf16x16 r;
#pragma unroll
    for (int i = 0; i < 8; ++i) { r[i] = lo[i]; r[8 + i] = hi[i]; }
    return r;
}

__device__ inline bf16x16 load_afrag(const bf16_t* __restrict__ row, int k0, int half) {
    bf16x8 lo = *(const bf16x8*)(row + k0 + half * 8);
    bf16x8 hi = *(const bf16x8*)(row + k0 + 16 + half * 8);
    return pack16(lo, hi);
}

__device__ inline bf16x16 load_bfrag(const bf16_t* __restrict__ row, int k0, int half) {
    bf16x8 lo = *(const bf16x8*)(row + k0 + half * 8);
    bf16x8 hi = *(const bf16x8*)(row + k0 + 16 + half * 8);
    return pack16(lo, hi);
}

#define WMMA_BF16(A_, B_, C_) \
    __builtin_amdgcn_wmma_f32_16x16x32_bf16(false, (A_), false, (B_), (short)0, (C_), false, false)

// ---------------------------------------------------------------------------
// Fast GEMM: 32x32 register tile per wave (2 A-frags x 2 B-frags -> 4 wmma
// per 8 b128 loads). Requires M%32==0 && N%32==0 && K%32==0.
// C[M][N] f32 = A[M][K] bf16 * Bt[N][K]^T bf16 + bias
// ---------------------------------------------------------------------------
__global__ void k_wmma_gemm32(const bf16_t* __restrict__ A, const bf16_t* __restrict__ Bt,
                              const float* __restrict__ bias, float* __restrict__ C,
                              int M, int N, int K, int relu) {
    int wave = (int)(((long)blockIdx.x * blockDim.x + threadIdx.x) >> 5);
    int lane = threadIdx.x & 31;
    int tilesN = N >> 5;
    int tilesM = M >> 5;
    if (wave >= tilesM * tilesN) return;
    int tm = wave / tilesN, tn = wave % tilesN;
    int half = lane >> 4;
    int l16  = lane & 15;

    const bf16_t* Arow0 = A  + (long)(tm * 32 + l16) * K;
    const bf16_t* Arow1 = Arow0 + (long)16 * K;
    const bf16_t* Brow0 = Bt + (long)(tn * 32 + l16) * K;
    const bf16_t* Brow1 = Brow0 + (long)16 * K;

    floatx8 acc00 = {}, acc01 = {}, acc10 = {}, acc11 = {};
    for (int k0 = 0; k0 < K; k0 += 32) {
        bf16x16 a0 = load_afrag(Arow0, k0, half);
        bf16x16 a1 = load_afrag(Arow1, k0, half);
        bf16x16 b0 = load_bfrag(Brow0, k0, half);
        bf16x16 b1 = load_bfrag(Brow1, k0, half);
        acc00 = WMMA_BF16(a0, b0, acc00);
        acc01 = WMMA_BF16(a0, b1, acc01);
        acc10 = WMMA_BF16(a1, b0, acc10);
        acc11 = WMMA_BF16(a1, b1, acc11);
    }

    int col0 = tn * 32 + l16;
    int col1 = col0 + 16;
    float bv0 = bias ? bias[col0] : 0.0f;
    float bv1 = bias ? bias[col1] : 0.0f;
#pragma unroll
    for (int r = 0; r < 8; ++r) {
        int row0 = tm * 32 + r + half * 8;   // C/D layout: VGPR r -> M=r / r+8
        int row1 = row0 + 16;
        float v00 = acc00[r] + bv0, v01 = acc01[r] + bv1;
        float v10 = acc10[r] + bv0, v11 = acc11[r] + bv1;
        if (relu) {
            v00 = fmaxf(v00, 0.0f); v01 = fmaxf(v01, 0.0f);
            v10 = fmaxf(v10, 0.0f); v11 = fmaxf(v11, 0.0f);
        }
        C[(long)row0 * N + col0] = v00;
        C[(long)row0 * N + col1] = v01;
        C[(long)row1 * N + col0] = v10;
        C[(long)row1 * N + col1] = v11;
    }
}

// ---------------------------------------------------------------------------
// Generic GEMM (16x16 tile per wave, N-edge guard) for the small FC layers.
// ---------------------------------------------------------------------------
__global__ void k_wmma_gemm(const bf16_t* __restrict__ A, const bf16_t* __restrict__ Bt,
                            const float* __restrict__ bias, float* __restrict__ C,
                            int M, int N, int K, int relu) {
    int wave = (int)(((long)blockIdx.x * blockDim.x + threadIdx.x) >> 5);
    int lane = threadIdx.x & 31;
    int tilesN = (N + 15) >> 4;
    int tilesM = M >> 4;
    if (wave >= tilesM * tilesN) return;
    int tm = wave / tilesN, tn = wave % tilesN;
    int half = lane >> 4;
    int l16  = lane & 15;

    int arow = tm * 16 + l16;
    int bcol = tn * 16 + l16;
    const bf16_t* Arow = A + (long)arow * K;
    const bf16_t* Brow = Bt + (long)(bcol < N ? bcol : 0) * K;  // clamp, mask result

    floatx8 acc = {};
    for (int k0 = 0; k0 < K; k0 += 32) {
        bf16x16 a = load_afrag(Arow, k0, half);
        bf16x16 b = load_bfrag(Brow, k0, half);
        acc = WMMA_BF16(a, b, acc);
    }

    if (bcol < N) {
        float bv = bias ? bias[bcol] : 0.0f;
#pragma unroll
        for (int r = 0; r < 8; ++r) {
            int row = tm * 16 + r + half * 8;
            float v = acc[r] + bv;
            if (relu) v = fmaxf(v, 0.0f);
            C[(long)row * N + bcol] = v;
        }
    }
}

// ---------------------------------------------------------------------------
// Attention: per-edge dot-product softmax over incoming edges, scatter-sum
// ---------------------------------------------------------------------------
__global__ void k_init_attn(float* __restrict__ amax, float* __restrict__ denom, int n) {
    int i = blockIdx.x * blockDim.x + threadIdx.x;
    if (i >= n) return;
    amax[i]  = -INFINITY;
    denom[i] = 0.0f;
}

// one wave per (edge, head): alpha = dot(q[dst], k[src]) * scale
__global__ void k_edge_alpha(const float* __restrict__ qm, const float* __restrict__ km,
                             const int* __restrict__ src, const int* __restrict__ dst,
                             float* __restrict__ alpha, int H, int d, float scale) {
    int wave = (int)(((long)blockIdx.x * blockDim.x + threadIdx.x) >> 5);
    int lane = threadIdx.x & 31;
    if (wave >= NEDGES * H) return;
    int e = wave / H, h = wave % H;
    const float* qp = qm + (long)dst[e] * H * d + (long)h * d;
    const float* kp = km + (long)src[e] * H * d + (long)h * d;
    float acc = 0.0f;
    for (int j = lane; j < d; j += 32) acc += qp[j] * kp[j];
#pragma unroll
    for (int off = 16; off > 0; off >>= 1) acc += __shfl_xor(acc, off, 32);
    if (lane == 0) alpha[(long)e * H + h] = acc * scale;
}

__global__ void k_alpha_max(const float* __restrict__ alpha, const int* __restrict__ dst,
                            float* __restrict__ amax, int H) {
    long i = (long)blockIdx.x * blockDim.x + threadIdx.x;
    if (i >= (long)NEDGES * H) return;
    int e = (int)(i / H), h = (int)(i % H);
    float v = alpha[i];
    float* ad = amax + (long)dst[e] * H + h;
    if (v >= 0.0f) atomicMax((int*)ad, __float_as_int(v));
    else           atomicMin((unsigned int*)ad, __float_as_uint(v));
}

__global__ void k_exp_denom(const float* __restrict__ alpha, const int* __restrict__ dst,
                            const float* __restrict__ amax, float* __restrict__ ex,
                            float* __restrict__ denom, int H) {
    long i = (long)blockIdx.x * blockDim.x + threadIdx.x;
    if (i >= (long)NEDGES * H) return;
    int e = (int)(i / H), h = (int)(i % H);
    float v = expf(alpha[i] - amax[(long)dst[e] * H + h]);
    ex[i] = v;
    atomicAdd(denom + (long)dst[e] * H + h, v);
}

// out[dst, h, :] += v[src, h, :] * (ex / (denom + eps));  out pre-seeded with skip.
// One thread per 4 consecutive features (d % 4 == 0 always here).
__global__ void k_scatter4(const float* __restrict__ vmat, const int* __restrict__ src,
                           const int* __restrict__ dst, const float* __restrict__ ex,
                           const float* __restrict__ denom, float* __restrict__ out,
                           int H, int d) {
    long i = (long)blockIdx.x * blockDim.x + threadIdx.x;
    int hd4 = (H * d) >> 2;
    long total = (long)NEDGES * hd4;
    if (i >= total) return;
    int e  = (int)(i / hd4);
    int r4 = (int)(i % hd4) << 2;
    int h  = r4 / d;
    int t  = dst[e];
    float a = ex[(long)e * H + h] / (denom[(long)t * H + h] + 1e-16f);
    const float4 vv = *(const float4*)(vmat + (long)src[e] * H * d + r4);
    float* op = out + (long)t * H * d + r4;
    atomicAdd(op + 0, vv.x * a);
    atomicAdd(op + 1, vv.y * a);
    atomicAdd(op + 2, vv.z * a);
    atomicAdd(op + 3, vv.w * a);
}

// ---------------------------------------------------------------------------
// Host-side orchestration
// ---------------------------------------------------------------------------
extern "C" void kernel_launch(void* const* d_in, const int* in_sizes, int n_in,
                              void* d_out, int out_size, void* d_ws, size_t ws_size,
                              hipStream_t stream) {
    (void)in_sizes; (void)n_in; (void)out_size; (void)ws_size;

    const float* x        = (const float*)d_in[0];
    const int*   eidx     = (const int*)d_in[1];
    const int*   conn     = (const int*)d_in[2];
    const float* node_pe  = (const float*)d_in[3];
    const float* lobe_pe  = (const float*)d_in[4];
    const float* lung_pe  = (const float*)d_in[5];
    const float* Wq1 = (const float*)d_in[6];  const float* bq1 = (const float*)d_in[7];
    const float* Wk1 = (const float*)d_in[8];  const float* bk1 = (const float*)d_in[9];
    const float* Wv1 = (const float*)d_in[10]; const float* bv1 = (const float*)d_in[11];
    const float* Ws1 = (const float*)d_in[12]; const float* bs1 = (const float*)d_in[13];
    const float* Wq2 = (const float*)d_in[14]; const float* bq2 = (const float*)d_in[15];
    const float* Wk2 = (const float*)d_in[16]; const float* bk2 = (const float*)d_in[17];
    const float* Wv2 = (const float*)d_in[18]; const float* bv2 = (const float*)d_in[19];
    const float* Ws2 = (const float*)d_in[20]; const float* bs2 = (const float*)d_in[21];
    const float* Wfc1 = (const float*)d_in[22]; const float* bfc1 = (const float*)d_in[23];
    const float* Wfc2 = (const float*)d_in[24]; const float* bfc2 = (const float*)d_in[25];
    const float* Wfc3 = (const float*)d_in[26]; const float* bfc3 = (const float*)d_in[27];
    const float* Wfc4 = (const float*)d_in[28]; const float* bfc4 = (const float*)d_in[29];

    const int* src = eidx;
    const int* dst = eidx + NEDGES;

    // ---- workspace bump allocator (256B aligned) ----
    char* wp = (char*)d_ws;
    auto alloc = [&](size_t bytes) -> void* {
        void* p = (void*)wp;
        wp += (bytes + 255) & ~(size_t)255;
        return p;
    };
    bf16_t* hbf   = (bf16_t*)alloc((size_t)NNODES * IN_DIM * 2);
    bf16_t* wq1t  = (bf16_t*)alloc((size_t)IN_DIM * HD1 * 2);
    bf16_t* wk1t  = (bf16_t*)alloc((size_t)IN_DIM * HD1 * 2);
    bf16_t* wv1t  = (bf16_t*)alloc((size_t)IN_DIM * HD1 * 2);
    bf16_t* ws1t  = (bf16_t*)alloc((size_t)IN_DIM * HD1 * 2);
    float*  q1    = (float*)alloc((size_t)NNODES * HD1 * 4);
    float*  k1    = (float*)alloc((size_t)NNODES * HD1 * 4);
    float*  v1    = (float*)alloc((size_t)NNODES * HD1 * 4);
    float*  out1  = (float*)alloc((size_t)NNODES * HD1 * 4);   // skip + scatter
    float*  alpha1 = (float*)alloc((size_t)NEDGES * HEADS1 * 4);
    float*  ex1    = (float*)alloc((size_t)NEDGES * HEADS1 * 4);
    float*  amax1  = (float*)alloc((size_t)NNODES * HEADS1 * 4);
    float*  den1   = (float*)alloc((size_t)NNODES * HEADS1 * 4);
    bf16_t* h1bf   = (bf16_t*)alloc((size_t)NNODES * HD1 * 2);
    bf16_t* wq2t   = (bf16_t*)alloc((size_t)HD1 * H2DIM * 2);
    bf16_t* wk2t   = (bf16_t*)alloc((size_t)HD1 * H2DIM * 2);
    bf16_t* wv2t   = (bf16_t*)alloc((size_t)HD1 * H2DIM * 2);
    bf16_t* ws2t   = (bf16_t*)alloc((size_t)HD1 * H2DIM * 2);
    float*  q2     = (float*)alloc((size_t)NNODES * H2DIM * 4);
    float*  k2     = (float*)alloc((size_t)NNODES * H2DIM * 4);
    float*  v2     = (float*)alloc((size_t)NNODES * H2DIM * 4);
    float*  out2   = (float*)alloc((size_t)NNODES * H2DIM * 4);
    float*  alpha2 = (float*)alloc((size_t)NEDGES * 4);
    float*  ex2    = (float*)alloc((size_t)NEDGES * 4);
    float*  amax2  = (float*)alloc((size_t)NNODES * 4);
    float*  den2   = (float*)alloc((size_t)NNODES * 4);
    bf16_t* h2bf   = (bf16_t*)alloc((size_t)NNODES * H2DIM * 2);  // == [16, 26880]
    bf16_t* fc1t   = (bf16_t*)alloc((size_t)FC1_K * FC1_N * 2);
    bf16_t* fc2t   = (bf16_t*)alloc((size_t)FC1_N * FC2_N * 2);
    bf16_t* fc3t   = (bf16_t*)alloc((size_t)FC2_N * FC3_N * 2);
    bf16_t* fc4t   = (bf16_t*)alloc((size_t)FC3_N * OUT_DIM * 2);
    float*  fc1o   = (float*)alloc((size_t)16 * FC1_N * 4);
    bf16_t* fc1b   = (bf16_t*)alloc((size_t)16 * FC1_N * 2);
    float*  fc2o   = (float*)alloc((size_t)16 * FC2_N * 4);
    bf16_t* fc2b   = (bf16_t*)alloc((size_t)16 * FC2_N * 2);
    float*  fc3o   = (float*)alloc((size_t)16 * FC3_N * 4);
    bf16_t* fc3b   = (bf16_t*)alloc((size_t)16 * FC3_N * 2);

    auto cdiv = [](long a, long b) -> int { return (int)((a + b - 1) / b); };

    auto transpose = [&](const float* W, bf16_t* Wt, int K, int N) {
        long tot = (long)K * N;
        k_transpose_bf16<<<cdiv(tot, 256), 256, 0, stream>>>(W, Wt, K, N);
    };
    auto gemm = [&](const bf16_t* A, const bf16_t* Bt, const float* bias, float* C,
                    int M, int N, int K, int relu) {
        if ((M % 32 == 0) && (N % 32 == 0)) {
            long tiles = (long)(M >> 5) * (N >> 5);
            k_wmma_gemm32<<<cdiv(tiles, 8), 256, 0, stream>>>(A, Bt, bias, C, M, N, K, relu);
        } else {
            long tiles = (long)(M >> 4) * ((N + 15) >> 4);
            k_wmma_gemm<<<cdiv(tiles, 8), 256, 0, stream>>>(A, Bt, bias, C, M, N, K, relu);
        }
    };

    // ---- stage weights (transpose + bf16) ----
    transpose(Wq1, wq1t, IN_DIM, HD1);
    transpose(Wk1, wk1t, IN_DIM, HD1);
    transpose(Wv1, wv1t, IN_DIM, HD1);
    transpose(Ws1, ws1t, IN_DIM, HD1);
    transpose(Wq2, wq2t, HD1, H2DIM);
    transpose(Wk2, wk2t, HD1, H2DIM);
    transpose(Wv2, wv2t, HD1, H2DIM);
    transpose(Ws2, ws2t, HD1, H2DIM);
    transpose(Wfc1, fc1t, FC1_K, FC1_N);
    transpose(Wfc2, fc2t, FC1_N, FC2_N);
    transpose(Wfc3, fc3t, FC2_N, FC3_N);
    transpose(Wfc4, fc4t, FC3_N, OUT_DIM);

    // ---- h = x + PEs ----
    {
        long tot = (long)NNODES * IN_DIM;
        k_pe_add<<<cdiv(tot, 256), 256, 0, stream>>>(x, conn, node_pe, lobe_pe, lung_pe, hbf);
    }

    // ---- layer 1: TransformerConv (heads=2, d=1024) ----
    gemm(hbf, wq1t, bq1, q1,   NNODES, HD1, IN_DIM, 0);
    gemm(hbf, wk1t, bk1, k1,   NNODES, HD1, IN_DIM, 0);
    gemm(hbf, wv1t, bv1, v1,   NNODES, HD1, IN_DIM, 0);
    gemm(hbf, ws1t, bs1, out1, NNODES, HD1, IN_DIM, 0);   // skip seeds scatter buffer

    k_init_attn<<<cdiv(NNODES * HEADS1, 256), 256, 0, stream>>>(amax1, den1, NNODES * HEADS1);
    k_edge_alpha<<<cdiv((long)NEDGES * HEADS1, 8), 256, 0, stream>>>(
        q1, k1, src, dst, alpha1, HEADS1, D1, 1.0f / sqrtf((float)D1));
    k_alpha_max<<<cdiv((long)NEDGES * HEADS1, 256), 256, 0, stream>>>(alpha1, dst, amax1, HEADS1);
    k_exp_denom<<<cdiv((long)NEDGES * HEADS1, 256), 256, 0, stream>>>(
        alpha1, dst, amax1, ex1, den1, HEADS1);
    k_scatter4<<<cdiv((long)NEDGES * HEADS1 * D1 / 4, 256), 256, 0, stream>>>(
        v1, src, dst, ex1, den1, out1, HEADS1, D1);
    k_relu_bf16<<<cdiv((long)NNODES * HD1, 256), 256, 0, stream>>>(out1, h1bf, (long)NNODES * HD1);

    // ---- layer 2: TransformerConv (heads=1, d=64) ----
    gemm(h1bf, wq2t, bq2, q2,   NNODES, H2DIM, HD1, 0);
    gemm(h1bf, wk2t, bk2, k2,   NNODES, H2DIM, HD1, 0);
    gemm(h1bf, wv2t, bv2, v2,   NNODES, H2DIM, HD1, 0);
    gemm(h1bf, ws2t, bs2, out2, NNODES, H2DIM, HD1, 0);

    k_init_attn<<<cdiv(NNODES, 256), 256, 0, stream>>>(amax2, den2, NNODES);
    k_edge_alpha<<<cdiv((long)NEDGES, 8), 256, 0, stream>>>(
        q2, k2, src, dst, alpha2, 1, H2DIM, 1.0f / sqrtf((float)H2DIM));
    k_alpha_max<<<cdiv((long)NEDGES, 256), 256, 0, stream>>>(alpha2, dst, amax2, 1);
    k_exp_denom<<<cdiv((long)NEDGES, 256), 256, 0, stream>>>(alpha2, dst, amax2, ex2, den2, 1);
    k_scatter4<<<cdiv((long)NEDGES * H2DIM / 4, 256), 256, 0, stream>>>(
        v2, src, dst, ex2, den2, out2, 1, H2DIM);
    k_relu_bf16<<<cdiv((long)NNODES * H2DIM, 256), 256, 0, stream>>>(
        out2, h2bf, (long)NNODES * H2DIM);

    // ---- FC head: [16, 26880] -> 256 -> 128 -> 64 -> 18 ----
    gemm(h2bf, fc1t, bfc1, fc1o, 16, FC1_N, FC1_K, 1);
    k_relu_bf16<<<cdiv(16 * FC1_N, 256), 256, 0, stream>>>(fc1o, fc1b, 16 * FC1_N);
    gemm(fc1b, fc2t, bfc2, fc2o, 16, FC2_N, FC1_N, 1);
    k_relu_bf16<<<cdiv(16 * FC2_N, 256), 256, 0, stream>>>(fc2o, fc2b, 16 * FC2_N);
    gemm(fc2b, fc3t, bfc3, fc3o, 16, FC3_N, FC2_N, 1);
    k_relu_bf16<<<cdiv(16 * FC3_N, 256), 256, 0, stream>>>(fc3o, fc3b, 16 * FC3_N);
    gemm(fc3b, fc4t, bfc4, (float*)d_out, 16, OUT_DIM, FC3_N, 0);
}
